// BaseSAE_35622458753216
// MI455X (gfx1250) — matrix operand order repaired and checked by reading, not compiled, not run on hardware
//
#include <hip/hip_runtime.h>
#include <hip/hip_bf16.h>

// ---------------------------------------------------------------------------
// TopK SAE forward for gfx1250 (MI455X)
//   z    = relu(x @ W^T + b_enc)          [8192, 16384]  (then top-64 sparsified)
//   xhat = z_sparse @ W + b_dec           [8192, 2048]
// d_out = [z | xhat] flat, float32.
// ---------------------------------------------------------------------------

#define BATCH 8192
#define D_IN  2048
#define WIDTH 16384
#define TOPK  64

#define TILE_M 128
#define TILE_N 128
#define TILE_K 32
#define LDSK   40          // padded K-stride (bf16 elems) -> 80B rows, 16B aligned

typedef __attribute__((ext_vector_type(16))) __bf16 v16bf;
typedef __attribute__((ext_vector_type(8)))  float  v8f;

__device__ __forceinline__ unsigned short f2bf(float f) {
    unsigned u = __float_as_uint(f);
    unsigned r = u + 0x7FFFu + ((u >> 16) & 1u);   // round-to-nearest-even
    return (unsigned short)(r >> 16);
}
__device__ __forceinline__ unsigned pack2(float a, float b) {
    return (unsigned)f2bf(a) | ((unsigned)f2bf(b) << 16);
}

union FragAB { uint4 u[2]; v16bf v; };

// ---------------------------------------------------------------------------
// Kernel 1: encode GEMM  z = relu(x @ W^T + b_enc), dense output
// ---------------------------------------------------------------------------
__global__ __launch_bounds__(256, 2)
void sae_encode(const float* __restrict__ x, const float* __restrict__ W,
                const float* __restrict__ benc, float* __restrict__ z)
{
    __shared__ unsigned short xs[TILE_M * LDSK];   // A tile, bf16, K-contiguous
    __shared__ unsigned short ws[TILE_N * LDSK];   // B tile, bf16, K-contiguous (W rows)

    const int tid  = threadIdx.x;
    const int lane = tid & 31;
    const int wave = tid >> 5;          // 8 waves
    const int wm   = wave & 3;          // 4 wave-rows  (32 rows each)
    const int wn   = wave >> 2;         // 2 wave-cols  (64 cols each)
    const int m0   = blockIdx.x * TILE_M;
    const int n0   = blockIdx.y * TILE_N;

    v8f acc[2][4];
    #pragma unroll
    for (int i = 0; i < 2; ++i)
        #pragma unroll
        for (int j = 0; j < 4; ++j)
            acc[i][j] = (v8f){0.f,0.f,0.f,0.f,0.f,0.f,0.f,0.f};

    const int lrow = tid >> 3;          // 0..31, row within a 32-row pass
    const int lc4  = (tid & 7) * 4;     // float col 0,4,...,28

    for (int k0 = 0; k0 < D_IN; k0 += TILE_K) {
        // ---- stage x/W tiles into LDS as bf16 (4 passes of 32 rows each) ----
        #pragma unroll
        for (int p = 0; p < 4; ++p) {
            int r = lrow + p * 32;
            float4 vx = *(const float4*)&x[(size_t)(m0 + r) * D_IN + k0 + lc4];
            float4 vw = *(const float4*)&W[(size_t)(n0 + r) * D_IN + k0 + lc4];
            *(uint2*)&xs[r * LDSK + lc4] = make_uint2(pack2(vx.x, vx.y), pack2(vx.z, vx.w));
            *(uint2*)&ws[r * LDSK + lc4] = make_uint2(pack2(vw.x, vw.y), pack2(vw.z, vw.w));
        }
        __syncthreads();

        // ---- build fragments (ISA 16-bit A/B VGPR layouts, K-contiguous) ----
        v16bf a[2], b[4];
        {
            const int arow = wm * 32 + (lane & 15);
            const int kh   = (lane >> 4) * 8;      // A: K 0..7|16..23 vs 8..15|24..31
            #pragma unroll
            for (int i = 0; i < 2; ++i) {
                const unsigned short* pa = &xs[(arow + i * 16) * LDSK];
                FragAB f;
                f.u[0] = *(const uint4*)&pa[kh];
                f.u[1] = *(const uint4*)&pa[kh + 16];
                a[i] = f.v;
            }
            const int brow = wn * 64 + (lane & 15);
            const int kb   = (lane >> 4) * 8;      // B: K 0..7|16..23 vs 8..15|24..31
            #pragma unroll
            for (int j = 0; j < 4; ++j) {
                const unsigned short* pb = &ws[(brow + j * 16) * LDSK];
                FragAB f;
                f.u[0] = *(const uint4*)&pb[kb];
                f.u[1] = *(const uint4*)&pb[kb + 16];
                b[j] = f.v;
            }
        }

        // ---- 8 WMMAs per wave per K-chunk ----
        #pragma unroll
        for (int i = 0; i < 2; ++i)
            #pragma unroll
            for (int j = 0; j < 4; ++j)
                acc[i][j] = __builtin_amdgcn_wmma_f32_16x16x32_bf16(
                    false, a[i], false, b[j], (short)0, acc[i][j], false, false);

        __syncthreads();
    }

    // ---- epilogue: + b_enc, relu, store dense z ----
    #pragma unroll
    for (int j = 0; j < 4; ++j) {
        const int n = n0 + wn * 64 + j * 16 + (lane & 15);
        const float bias = benc[n];
        #pragma unroll
        for (int i = 0; i < 2; ++i) {
            const int mb = m0 + wm * 32 + i * 16 + (lane >> 4) * 8;
            #pragma unroll
            for (int r = 0; r < 8; ++r) {
                float v = acc[i][j][r] + bias;
                v = v > 0.f ? v : 0.f;
                z[(size_t)(mb + r) * WIDTH + n] = v;
            }
        }
    }
}

// ---------------------------------------------------------------------------
// Kernel 2: exact top-64 per row via 4-pass radix select on float bits
// (relu output >= 0, so uint order == float order). Deterministic,
// index-ordered tie-breaking. Emits idx[64], val[64] per row into ws.
// ---------------------------------------------------------------------------
__global__ __launch_bounds__(256)
void sae_topk(const float* __restrict__ z, int* __restrict__ idxOut,
              float* __restrict__ valOut)
{
    extern __shared__ float srow[];                 // 16384 floats = 64 KB
    __shared__ unsigned hist[256];
    __shared__ unsigned gcnt[256], ecnt[256];
    __shared__ unsigned sPrefVal, sPrefMask, sNeed;

    const int b   = blockIdx.x;
    const int tid = threadIdx.x;
    const float* zr = &z[(size_t)b * WIDTH];

    // coalesced row load into LDS
    #pragma unroll
    for (int i = 0; i < 16; ++i)
        ((float4*)srow)[tid + i * 256] = ((const float4*)zr)[tid + i * 256];

    if (tid == 0) { sPrefVal = 0u; sPrefMask = 0u; sNeed = TOPK; }
    __syncthreads();

    const float* my = &srow[tid * 64];              // contiguous 64-elem chunk

    for (int shift = 24; shift >= 0; shift -= 8) {
        hist[tid] = 0u;
        __syncthreads();
        const unsigned pv = sPrefVal, pm = sPrefMask;
        for (int j = 0; j < 64; ++j) {
            int jj = (j + tid) & 63;                // stagger: kill bank conflicts
            unsigned v = __float_as_uint(my[jj]);
            if ((v & pm) == pv) atomicAdd(&hist[(v >> shift) & 255u], 1u);
        }
        __syncthreads();
        if (tid == 0) {
            unsigned need = sNeed, above = 0; int bkt = 255;
            for (; bkt >= 0; --bkt) {
                unsigned c = hist[bkt];
                if (above + c >= need) break;
                above += c;
            }
            if (bkt < 0) bkt = 0;
            sPrefVal  = pv | ((unsigned)bkt << shift);
            sPrefMask = pm | (0xFFu << shift);
            sNeed     = need - above;               // equals still needed
        }
        __syncthreads();
    }

    const unsigned T      = sPrefVal;               // exact k-th value (bits)
    const unsigned needEq = sNeed;                  // #ties to keep
    const unsigned G      = TOPK - needEq;          // #strictly greater

    unsigned g = 0, e = 0;
    for (int j = 0; j < 64; ++j) {
        unsigned v = __float_as_uint(my[j]);
        if (v > T) ++g; else if (v == T) ++e;
    }
    gcnt[tid] = g; ecnt[tid] = e;
    __syncthreads();
    if (tid == 0) {                                 // serial exclusive scans
        unsigned ag = 0, ae = 0;
        for (int t = 0; t < 256; ++t) {
            unsigned tg = gcnt[t], te = ecnt[t];
            gcnt[t] = ag; ecnt[t] = ae;
            ag += tg; ae += te;
        }
    }
    __syncthreads();

    unsigned gs = gcnt[tid], es = ecnt[tid];
    int*   oi = &idxOut[(size_t)b * TOPK];
    float* ov = &valOut[(size_t)b * TOPK];
    for (int j = 0; j < 64; ++j) {
        float    f = my[j];
        unsigned v = __float_as_uint(f);
        int idx = tid * 64 + j;
        if (v > T)      { unsigned s = gs++; oi[s] = idx; ov[s] = f; }
        else if (v == T){ unsigned r = es++;
                          if (r < needEq) { unsigned s = G + r; oi[s] = idx; ov[s] = f; } }
    }
}

// ---------------------------------------------------------------------------
// Kernel 3: rewrite z rows as sparse (zero row, scatter top-64)
// ---------------------------------------------------------------------------
__global__ __launch_bounds__(256)
void sae_scatter(const int* __restrict__ idx, const float* __restrict__ vals,
                 float* __restrict__ z)
{
    const int b = blockIdx.x;
    float4* zr = (float4*)&z[(size_t)b * WIDTH];
    const float4 zero = make_float4(0.f, 0.f, 0.f, 0.f);
    #pragma unroll
    for (int i = 0; i < 16; ++i) zr[threadIdx.x + i * 256] = zero;
    __syncthreads();
    if (threadIdx.x < TOPK) {
        int j = threadIdx.x;
        z[(size_t)b * WIDTH + idx[(size_t)b * TOPK + j]] = vals[(size_t)b * TOPK + j];
    }
}

// ---------------------------------------------------------------------------
// Kernel 4: decode  xhat = sum_j val_j * W[idx_j, :] + b_dec   (W L2-resident)
// ---------------------------------------------------------------------------
__global__ __launch_bounds__(256)
void sae_decode(const float* __restrict__ W, const float* __restrict__ bdec,
                const int* __restrict__ idx, const float* __restrict__ vals,
                float* __restrict__ xhat)
{
    __shared__ int   sIdx[TOPK];
    __shared__ float sVal[TOPK];
    const int b = blockIdx.x;
    if (threadIdx.x < TOPK) {
        sIdx[threadIdx.x] = idx[(size_t)b * TOPK + threadIdx.x];
        sVal[threadIdx.x] = vals[(size_t)b * TOPK + threadIdx.x];
    }
    __syncthreads();

    const int col = threadIdx.x * 8;                // 256 thr * 8 = 2048
    float a0 = bdec[col+0], a1 = bdec[col+1], a2 = bdec[col+2], a3 = bdec[col+3];
    float a4 = bdec[col+4], a5 = bdec[col+5], a6 = bdec[col+6], a7 = bdec[col+7];

    for (int j = 0; j < TOPK; ++j) {
        const float s = sVal[j];
        const float* wr = &W[(size_t)sIdx[j] * D_IN + col];
        float4 w0 = *(const float4*)wr;
        float4 w1 = *(const float4*)(wr + 4);
        a0 += s * w0.x; a1 += s * w0.y; a2 += s * w0.z; a3 += s * w0.w;
        a4 += s * w1.x; a5 += s * w1.y; a6 += s * w1.z; a7 += s * w1.w;
    }
    float* out = &xhat[(size_t)b * D_IN + col];
    out[0]=a0; out[1]=a1; out[2]=a2; out[3]=a3;
    out[4]=a4; out[5]=a5; out[6]=a6; out[7]=a7;
}

// ---------------------------------------------------------------------------
extern "C" void kernel_launch(void* const* d_in, const int* in_sizes, int n_in,
                              void* d_out, int out_size, void* d_ws, size_t ws_size,
                              hipStream_t stream)
{
    (void)in_sizes; (void)n_in; (void)out_size; (void)ws_size;

    const float* x    = (const float*)d_in[0];   // [8192, 2048]
    const float* W    = (const float*)d_in[1];   // [16384, 2048]
    const float* benc = (const float*)d_in[2];   // [16384]
    const float* bdec = (const float*)d_in[3];   // [2048]
    // d_in[4] = topk (compile-time constant 64 here)

    float* z    = (float*)d_out;                       // [8192, 16384]
    float* xhat = z + (size_t)BATCH * WIDTH;           // [8192, 2048]

    int*   idx  = (int*)d_ws;                          // [8192, 64]  (2 MB)
    float* vals = (float*)((char*)d_ws + (size_t)BATCH * TOPK * sizeof(int)); // 2 MB

    dim3 egrid(BATCH / TILE_M, WIDTH / TILE_N);        // (64, 128): x-major -> W L2 reuse
    sae_encode <<<egrid, 256, 0, stream>>>(x, W, benc, z);
    sae_topk   <<<BATCH, 256, WIDTH * sizeof(float), stream>>>(z, idx, vals);
    sae_scatter<<<BATCH, 256, 0, stream>>>(idx, vals, z);
    sae_decode <<<BATCH, 256, 0, stream>>>(W, bdec, idx, vals, xhat);
}
